// SPAIRPointPosFlow_15470472200214
// MI455X (gfx1250) — matrix operand order repaired and checked by reading, not compiled, not run on hardware
//
#include <hip/hip_runtime.h>
#include <hip/hip_bf16.h>
#include <math.h>

typedef _Float16 f16;
typedef f16   v16h  __attribute__((ext_vector_type(16)));
typedef f16   half8 __attribute__((ext_vector_type(8)));
typedef float v8f   __attribute__((ext_vector_type(8)));
typedef int   v4i   __attribute__((vector_size(16)));

#define NTOT   15360
#define NBASE  12288
#define KNBR   64
#define NSC    8
#define CMAX   4096

__device__ __forceinline__ v8f wmma_f16(v16h a, v16h b, v8f c) {
  // D = A(16x32 f16) * B(32x16 f16) + C(16x16 f32)
  return __builtin_amdgcn_wmma_f32_16x16x32_f16(false, a, false, b, (short)0, c, false, false);
}

// ---- gfx1250 async copy: 16B global -> LDS, tracked on ASYNCcnt ----
__device__ __forceinline__ void async_copy16(f16* dst_lds, const f16* src_glob) {
#if __has_builtin(__builtin_amdgcn_global_load_async_to_lds_b128)
  __builtin_amdgcn_global_load_async_to_lds_b128(
      (__attribute__((address_space(1))) v4i*)(uintptr_t)src_glob,
      (__attribute__((address_space(3))) v4i*)(uintptr_t)(void*)dst_lds, 0, 0);
#else
  unsigned lds_off = (unsigned)(uintptr_t)(__attribute__((address_space(3))) f16*)dst_lds;
  asm volatile("global_load_async_to_lds_b128 %0, %1, off"
               :: "v"(lds_off), "v"((unsigned long long)(uintptr_t)src_glob)
               : "memory");
#endif
}

__device__ __forceinline__ void wait_async0() {
#if __has_builtin(__builtin_amdgcn_s_wait_asynccnt)
  __builtin_amdgcn_s_wait_asynccnt(0);
#else
  asm volatile("s_wait_asynccnt 0x0" ::: "memory");
#endif
}

// A fragment from row-major f16 [rows][stride]; stride multiple of 32.
__device__ __forceinline__ v16h load_a_frag(const f16* A, int stride, int mt, int kt, int lane) {
  int row  = mt * 16 + (lane & 15);
  int base = kt * 32 + ((lane & 16) ? 8 : 0);
  half8 lo = *(const half8*)(A + row * stride + base);
  half8 hi = *(const half8*)(A + row * stride + base + 16);
  v16h a;
  #pragma unroll
  for (int e = 0; e < 8; ++e) { a[e] = lo[e]; a[e + 8] = hi[e]; }
  return a;
}

// B fragment from transposed weight Wt[N_pad][K_pad] f16 (K contiguous per row).
__device__ __forceinline__ v16h load_b_frag(const f16* Wt, int Kp, int nt, int kt, int lane) {
  int n  = nt * 16 + (lane & 15);
  int k0 = kt * 32 + ((lane & 16) ? 8 : 0);
  const f16* p = Wt + (size_t)n * Kp + k0;
  half8 lo = *(const half8*)(p);
  half8 hi = *(const half8*)(p + 16);
  v16h b;
  #pragma unroll
  for (int e = 0; e < 8; ++e) { b[e] = lo[e]; b[e + 8] = hi[e]; }
  return b;
}

// ---------------- setup: batch_full, scene lists, zero accumulators ----------------
__global__ void setup_kernel(const int* __restrict__ batch, const int* __restrict__ sidx,
                             int* bfull, int* soff, int* scnt, int* slist,
                             float* sums, int* outI) {
  __shared__ int cntL[NSC], curL[NSC], offL[NSC + 1];
  int tid = threadIdx.x;
  if (tid < NSC) { cntL[tid] = 0; curL[tid] = 0; }
  if (tid < NSC * 3) sums[tid] = 0.f;
  __syncthreads();
  for (int n = tid; n < NTOT; n += 256) {
    int s = (n < NBASE) ? batch[n] : batch[sidx[n - NBASE]];
    bfull[n] = s;
    outI[n]  = s;                 // second tuple element of the output
    atomicAdd(&cntL[s], 1);
  }
  __syncthreads();
  if (tid == 0) {
    int acc = 0;
    for (int s = 0; s < NSC; ++s) { offL[s] = acc; acc += cntL[s]; scnt[s] = cntL[s]; }
    offL[NSC] = acc;
    for (int s = 0; s <= NSC; ++s) soff[s] = offL[s];
  }
  __syncthreads();
  for (int n = tid; n < NTOT; n += 256) {
    int s = bfull[n];
    int p = offL[s] + atomicAdd(&curL[s], 1);
    slist[p] = n;
  }
}

// ---------------- feat0 = concat(z[batch_full], noise) in f16, stride 136 ----------------
__global__ void feat0_kernel(const float* __restrict__ z, const float* __restrict__ noise,
                             const int* __restrict__ bfull, f16* __restrict__ featA) {
  int total = NTOT * 136;
  for (int i = blockIdx.x * 256 + threadIdx.x; i < total; i += gridDim.x * 256) {
    int n = i / 136, c = i - n * 136;   // constant divisor
    float v = (c < 128) ? z[bfull[n] * 128 + c] : (c < 131 ? noise[n * 3 + (c - 128)] : 0.f);
    featA[i] = (f16)v;
  }
}

// ---------------- fp32 (K,N) -> f16 transposed padded [Np][Kp] ----------------
__global__ void convw_kernel(const float* __restrict__ src, f16* __restrict__ dst,
                             int K, int N, int Kp, int Np) {
  int total = Kp * Np;
  for (int i = blockIdx.x * 256 + threadIdx.x; i < total; i += gridDim.x * 256) {
    int n = i / Kp, k = i - n * Kp;
    dst[i] = (k < K && n < N) ? (f16)src[(size_t)k * N + n] : (f16)0.f;
  }
}

// ---------------- radius graph + top-64 (block per node) ----------------
__global__ void radius_topk_kernel(const float* __restrict__ pos, const int* __restrict__ bfull,
                                   const int* __restrict__ slist, const int* __restrict__ soff,
                                   float r2, int* __restrict__ nbr, float* __restrict__ maskA) {
  __shared__ unsigned long long keys[CMAX];
  __shared__ float d2s[CMAX];
  __shared__ unsigned long long red[256];
  int node = blockIdx.x, tid = threadIdx.x;
  int s = bfull[node];
  int beg = soff[s];
  int cnt = soff[s + 1] - beg;
  if (cnt > CMAX) cnt = CMAX;
  float px = pos[node * 3], py = pos[node * 3 + 1], pz = pos[node * 3 + 2];
  for (int t = tid; t < cnt; t += 256) {
    int j = slist[beg + t];
    float dx = pos[j * 3] - px, dy = pos[j * 3 + 1] - py, dz = pos[j * 3 + 2] - pz;
    float d2 = dx * dx + dy * dy + dz * dz;
    d2s[t] = d2;
    unsigned u = __float_as_uint(d2);
    u = (u & 0x80000000u) ? ~u : (u | 0x80000000u);
    keys[t] = ((unsigned long long)u << 32) |
              ((unsigned long long)(j & 0xFFFF) << 16) | (unsigned)(t & 0xFFFF);
  }
  __syncthreads();
  for (int k = 0; k < KNBR; ++k) {
    unsigned long long best = ~0ULL;
    for (int t = tid; t < cnt; t += 256) { unsigned long long v = keys[t]; if (v < best) best = v; }
    red[tid] = best;
    __syncthreads();
    for (int off = 128; off > 0; off >>= 1) {
      if (tid < off && red[tid + off] < red[tid]) red[tid] = red[tid + off];
      __syncthreads();
    }
    if (tid == 0) {
      unsigned long long w = red[0];
      if (w == ~0ULL) { nbr[node * KNBR + k] = node; maskA[node * KNBR + k] = 0.f; }
      else {
        int slot = (int)(w & 0xFFFF);
        int j    = (int)((w >> 16) & 0xFFFF);
        nbr[node * KNBR + k]   = j;
        maskA[node * KNBR + k] = (d2s[slot] <= r2) ? 1.f : 0.f;
        keys[slot] = ~0ULL;
      }
    }
    __syncthreads();
  }
}

// ---------------- edge f-MLP (WMMA) + masked column max -> agg ----------------
// feat is f16 with row stride Dp8 (multiple of 8, zero-padded past D).
__global__ void fmlp_kernel(const f16* __restrict__ feat, int D, int Dp8,
                            const float* __restrict__ pos,
                            const int* __restrict__ nbr, const float* __restrict__ maskA,
                            const f16* __restrict__ W1t, int Epad, int FH,
                            const float* __restrict__ b1,
                            const f16* __restrict__ W2t, int FH2p,
                            int FO, int FOpad, const float* __restrict__ b2,
                            float* __restrict__ aggOut) {
  __shared__ alignas(16) f16 As[64 * 160];
  __shared__ alignas(16) f16 Hs[64 * 128];
  __shared__ float aggL[128];
  __shared__ float maskRow[64];
  int node = blockIdx.x, tid = threadIdx.x;
  int lane = tid & 31, wave = tid >> 5;
  if (tid == 0) { __builtin_prefetch(W1t, 0, 0); __builtin_prefetch(W2t, 0, 0); }
  int chunks = Dp8 >> 3;
  // async DMA gather: each wave owns 8 rows; each 4-lane group owns a row; ck strides by 4
  {
    int j  = wave * 8 + (lane >> 2);
    int nb = nbr[node * KNBR + j];
    const f16* srow = feat + (size_t)nb * Dp8;
    f16* drow = As + j * Epad;
    for (int ck = (lane & 3); ck < chunks; ck += 4)
      async_copy16(drow + ck * 8, srow + ck * 8);
  }
  // zero pad cols [Dp8, Epad): row = tid>>3 (+32), col = Dp8 + (tid&7) (+8)
  for (int j = tid >> 3; j < 64; j += 32) {
    f16* drow = As + j * Epad;
    for (int c = Dp8 + (tid & 7); c < Epad; c += 8) drow[c] = (f16)0.f;
  }
  if (tid < 64) maskRow[tid] = maskA[node * KNBR + tid];
  for (int i = tid; i < 64 * FH2p; i += 256) Hs[i] = (f16)0.f;
  if (tid < FOpad) ((int*)aggL)[tid] = 0xFF800000; // -inf
  wait_async0();
  __syncthreads();
  // rel = pos[nbr] - pos[i] into cols [D, D+3)
  if (tid < 192) {
    int j = tid / 3, c = tid - j * 3;   // constant divisor
    int nb = nbr[node * KNBR + j];
    As[j * Epad + D + c] = (f16)(pos[nb * 3 + c] - pos[node * 3 + c]);
  }
  __syncthreads();
  // H = relu(A @ W1 + b1)
  int nt1 = FH >> 4, T1 = 4 * nt1, kc1 = Epad >> 5;
  for (int t = wave; t < T1; t += 8) {
    int mt = t / nt1, nt = t - mt * nt1;
    v8f acc = {0.f, 0.f, 0.f, 0.f, 0.f, 0.f, 0.f, 0.f};
    for (int kt = 0; kt < kc1; ++kt)
      acc = wmma_f16(load_a_frag(As, Epad, mt, kt, lane),
                     load_b_frag(W1t, Epad, nt, kt, lane), acc);
    int n = nt * 16 + (lane & 15);
    float bias = b1[n];
    int mhi = (lane & 16) ? 8 : 0;
    #pragma unroll
    for (int r = 0; r < 8; ++r) {
      float v = acc[r] + bias; v = v > 0.f ? v : 0.f;
      Hs[(mt * 16 + r + mhi) * FH2p + n] = (f16)v;
    }
  }
  __syncthreads();
  // Y = relu(H @ W2 + b2); masked column max -> aggL
  int nt2 = FOpad >> 4, T2 = 4 * nt2, kc2 = FH2p >> 5;
  for (int t = wave; t < T2; t += 8) {
    int mt = t / nt2, nt = t - mt * nt2;
    v8f acc = {0.f, 0.f, 0.f, 0.f, 0.f, 0.f, 0.f, 0.f};
    for (int kt = 0; kt < kc2; ++kt)
      acc = wmma_f16(load_a_frag(Hs, FH2p, mt, kt, lane),
                     load_b_frag(W2t, FH2p, nt, kt, lane), acc);
    int n = nt * 16 + (lane & 15);
    int mhi = (lane & 16) ? 8 : 0;
    if (n < FO) {
      float bias = b2[n];
      #pragma unroll
      for (int r = 0; r < 8; ++r) {
        int row = mt * 16 + r + mhi;
        float v = acc[r] + bias; v = v > 0.f ? v : 0.f;
        if (maskRow[row] > 0.5f) atomicMax((int*)&aggL[n], __float_as_int(v));
      }
    }
  }
  __syncthreads();
  if (tid < FO)
    aggOut[(size_t)node * FO + tid] = __int_as_float(((int*)aggL)[tid]);
}

// ---------------- g-MLP (WMMA), 32 nodes per block, celu/split outputs ----------------
__global__ void gmlp_kernel(const f16* __restrict__ feat, int D, int Dp8,
                            const float* __restrict__ agg, int FO, int K1p,
                            const f16* __restrict__ GW1t, int GH, const float* __restrict__ gb1,
                            const f16* __restrict__ GW2t, int GH2p,
                            int GO, int GOpad, const float* __restrict__ gb2,
                            int S, f16* __restrict__ featOut, float* __restrict__ popOut) {
  __shared__ alignas(16) f16 Xs[32 * 288];
  __shared__ alignas(16) f16 Hs[32 * 128];
  __shared__ float Ys[32 * 80];
  int n0 = blockIdx.x * 32, tid = threadIdx.x;
  int lane = tid & 31, wave = tid >> 5;
  if (tid == 0) { __builtin_prefetch(GW1t, 0, 0); __builtin_prefetch(GW2t, 0, 0); }
  int K1 = D + FO;
  int chunks = Dp8 >> 3;
  // async DMA: each wave owns 4 rows; each 8-lane group owns a row; ck strides by 8
  {
    int j = wave * 4 + (lane >> 3);
    const f16* srow = feat + (size_t)(n0 + j) * Dp8;
    f16* drow = Xs + j * K1p;
    for (int ck = (lane & 7); ck < chunks; ck += 8)
      async_copy16(drow + ck * 8, srow + ck * 8);
  }
  // zero pad cols [K1, K1p): row = tid>>3, col strided by 8
  {
    int j = tid >> 3;
    f16* drow = Xs + j * K1p;
    for (int c = K1 + (tid & 7); c < K1p; c += 8) drow[c] = (f16)0.f;
  }
  for (int i = tid; i < 32 * GH2p; i += 256) Hs[i] = (f16)0.f;
  wait_async0();
  __syncthreads();
  // agg (fp32 -> f16) into cols [D, D+FO): row = tid>>3, col strided by 8
  {
    int j = tid >> 3;
    const float* arow = agg + (size_t)(n0 + j) * FO;
    f16* drow = Xs + j * K1p + D;
    for (int c = (tid & 7); c < FO; c += 8) drow[c] = (f16)arow[c];
  }
  __syncthreads();
  int nt1 = GH >> 4, T1 = 2 * nt1, kc1 = K1p >> 5;
  for (int t = wave; t < T1; t += 8) {
    int mt = t / nt1, nt = t - mt * nt1;
    v8f acc = {0.f, 0.f, 0.f, 0.f, 0.f, 0.f, 0.f, 0.f};
    for (int kt = 0; kt < kc1; ++kt)
      acc = wmma_f16(load_a_frag(Xs, K1p, mt, kt, lane),
                     load_b_frag(GW1t, K1p, nt, kt, lane), acc);
    int n = nt * 16 + (lane & 15);
    float bias = gb1[n];
    int mhi = (lane & 16) ? 8 : 0;
    #pragma unroll
    for (int r = 0; r < 8; ++r) {
      float v = acc[r] + bias; v = v > 0.f ? v : 0.f;
      Hs[(mt * 16 + r + mhi) * GH2p + n] = (f16)v;
    }
  }
  __syncthreads();
  int nt2 = GOpad >> 4, T2 = 2 * nt2, kc2 = GH2p >> 5;
  for (int t = wave; t < T2; t += 8) {
    int mt = t / nt2, nt = t - mt * nt2;
    v8f acc = {0.f, 0.f, 0.f, 0.f, 0.f, 0.f, 0.f, 0.f};
    for (int kt = 0; kt < kc2; ++kt)
      acc = wmma_f16(load_a_frag(Hs, GH2p, mt, kt, lane),
                     load_b_frag(GW2t, GH2p, nt, kt, lane), acc);
    int n = nt * 16 + (lane & 15);
    float bias = (n < GO) ? gb2[n] : 0.f;
    int mhi = (lane & 16) ? 8 : 0;
    #pragma unroll
    for (int r = 0; r < 8; ++r)
      Ys[(mt * 16 + r + mhi) * GOpad + n] = acc[r] + bias;
  }
  __syncthreads();
  // outputs: row = tid>>3, col strided by 8 over GO
  {
    int j = tid >> 3;
    int node = n0 + j;
    for (int c = (tid & 7); c < GO; c += 8) {
      float v = Ys[j * GOpad + c];
      if (c < S) featOut[(size_t)node * S + c] = (f16)((v > 0.f) ? v : (expf(v) - 1.f)); // celu
      else       popOut[(size_t)node * 3 + (c - S)] = v;
    }
  }
}

// ---------------- deterministic per-scene sums, then centroid subtract ----------------
__global__ void scene_sum_kernel(const float* __restrict__ pop, const int* __restrict__ bfull,
                                 float* __restrict__ sums) {
  __shared__ float rx[256], ry[256], rz[256];
  int s = blockIdx.x, tid = threadIdx.x;
  float ax = 0.f, ay = 0.f, az = 0.f;
  for (int n = tid; n < NTOT; n += 256)
    if (bfull[n] == s) { ax += pop[n * 3]; ay += pop[n * 3 + 1]; az += pop[n * 3 + 2]; }
  rx[tid] = ax; ry[tid] = ay; rz[tid] = az;
  __syncthreads();
  for (int off = 128; off > 0; off >>= 1) {
    if (tid < off) { rx[tid] += rx[tid + off]; ry[tid] += ry[tid + off]; rz[tid] += rz[tid + off]; }
    __syncthreads();
  }
  if (tid == 0) { sums[s * 3] = rx[0]; sums[s * 3 + 1] = ry[0]; sums[s * 3 + 2] = rz[0]; }
}

__global__ void finalize_kernel(const float* __restrict__ pop, const int* __restrict__ bfull,
                                const float* __restrict__ sums, const int* __restrict__ scnt,
                                float* __restrict__ out) {
  for (int n = blockIdx.x * 256 + threadIdx.x; n < NTOT; n += gridDim.x * 256) {
    int s = bfull[n];
    float c = (float)(scnt[s] > 1 ? scnt[s] : 1);
    out[n * 3]     = pop[n * 3]     - sums[s * 3]     / c;
    out[n * 3 + 1] = pop[n * 3 + 1] - sums[s * 3 + 1] / c;
    out[n * 3 + 2] = pop[n * 3 + 2] - sums[s * 3 + 2] / c;
  }
}

extern "C" void kernel_launch(void* const* d_in, const int* in_sizes, int n_in,
                              void* d_out, int out_size, void* d_ws, size_t ws_size,
                              hipStream_t stream) {
  const float* z     = (const float*)d_in[0];
  const int*   batch = (const int*)d_in[1];
  const int*   sidx  = (const int*)d_in[2];
  const float* noise = (const float*)d_in[3];

  char* W = (char*)d_ws;
  size_t o = 0;
  auto A = [&](size_t bytes) { size_t r = o; o += (bytes + 255) & ~(size_t)255; return r; };

  int*   bfull = (int*)  (W + A(NTOT * 4));
  int*   soff  = (int*)  (W + A(16 * 4));
  int*   scnt  = (int*)  (W + A(NSC * 4));
  int*   slist = (int*)  (W + A(NTOT * 4));
  float* sums  = (float*)(W + A(NSC * 3 * 4));
  int*   nbr   = (int*)  (W + A((size_t)NTOT * KNBR * 4));
  float* maskA = (float*)(W + A((size_t)NTOT * KNBR * 4));
  f16*   featA = (f16*)  (W + A((size_t)NTOT * 136 * 2));   // layer-1 feat, f16, stride 136
  f16*   featB = (f16*)  (W + A((size_t)NTOT * 64 * 2));    // layer-2 feat, f16, stride 64
  f16*   featC = (f16*)  (W + A((size_t)NTOT * 32 * 2));    // layer-3 feat, f16, stride 32
  float* popA  = (float*)(W + A((size_t)NTOT * 3 * 4));
  float* popB  = (float*)(W + A((size_t)NTOT * 3 * 4));
  float* aggB  = (float*)(W + A((size_t)NTOT * 128 * 4));

  // f16 transposed, padded weights: [Np][Kp]
  struct WS { int src, K, N, Kp, Np; f16* dst; };
  WS ws12[12] = {
    {4, 134, 128, 160, 128, nullptr}, {6, 128, 128, 128, 128, nullptr},
    {8, 259, 128, 288, 128, nullptr}, {10, 128, 67, 128, 80, nullptr},
    {12, 67, 64, 96, 64, nullptr},    {14, 64, 64, 64, 64, nullptr},
    {16, 128, 64, 128, 64, nullptr},  {18, 64, 35, 64, 48, nullptr},
    {20, 35, 16, 64, 16, nullptr},    {22, 16, 16, 32, 16, nullptr},
    {24, 48, 16, 64, 16, nullptr},    {26, 16, 3, 32, 16, nullptr},
  };
  for (int i = 0; i < 12; ++i)
    ws12[i].dst = (f16*)(W + A((size_t)ws12[i].Kp * ws12[i].Np * 2));

  int* outI = (int*)d_out + NTOT * 3;

  setup_kernel<<<1, 256, 0, stream>>>(batch, sidx, bfull, soff, scnt, slist, sums, outI);
  feat0_kernel<<<2048, 256, 0, stream>>>(z, noise, bfull, featA);
  for (int i = 0; i < 12; ++i) {
    int total = ws12[i].Kp * ws12[i].Np;
    int grid = (total + 255) / 256; if (grid > 512) grid = 512;
    convw_kernel<<<grid, 256, 0, stream>>>((const float*)d_in[ws12[i].src], ws12[i].dst,
                                           ws12[i].K, ws12[i].N, ws12[i].Kp, ws12[i].Np);
  }

  // ---------- layer 1: D=131 (stride 136), E=134->160, FH=128, FO=128; g: 259->288, GH=128, GO=67->80, S=64 ----------
  radius_topk_kernel<<<NTOT, 256, 0, stream>>>(noise, bfull, slist, soff, 0.2f * 0.2f, nbr, maskA);
  fmlp_kernel<<<NTOT, 256, 0, stream>>>(featA, 131, 136, noise, nbr, maskA,
                                        ws12[0].dst, 160, 128, (const float*)d_in[5],
                                        ws12[1].dst, 128, 128, 128, (const float*)d_in[7], aggB);
  gmlp_kernel<<<NTOT / 32, 256, 0, stream>>>(featA, 131, 136, aggB, 128, 288,
                                             ws12[2].dst, 128, (const float*)d_in[9],
                                             ws12[3].dst, 128, 67, 80, (const float*)d_in[11],
                                             64, featB, popA);

  // ---------- layer 2: D=64, E=67->96, FH=64, FO=64; g: 128, GH=64, GO=35->48, S=32 ----------
  radius_topk_kernel<<<NTOT, 256, 0, stream>>>(popA, bfull, slist, soff, 0.1f * 0.1f, nbr, maskA);
  fmlp_kernel<<<NTOT, 256, 0, stream>>>(featB, 64, 64, popA, nbr, maskA,
                                        ws12[4].dst, 96, 64, (const float*)d_in[13],
                                        ws12[5].dst, 64, 64, 64, (const float*)d_in[15], aggB);
  gmlp_kernel<<<NTOT / 32, 256, 0, stream>>>(featB, 64, 64, aggB, 64, 128,
                                             ws12[6].dst, 64, (const float*)d_in[17],
                                             ws12[7].dst, 64, 35, 48, (const float*)d_in[19],
                                             32, featC, popB);

  // ---------- layer 3: D=32, E=35->64, FH=16 (pad K->32), FO=16; g: 48->64, GH=16, GO=3->16, S=0 ----------
  radius_topk_kernel<<<NTOT, 256, 0, stream>>>(popB, bfull, slist, soff, 0.05f * 0.05f, nbr, maskA);
  fmlp_kernel<<<NTOT, 256, 0, stream>>>(featC, 32, 32, popB, nbr, maskA,
                                        ws12[8].dst, 64, 16, (const float*)d_in[21],
                                        ws12[9].dst, 32, 16, 16, (const float*)d_in[23], aggB);
  gmlp_kernel<<<NTOT / 32, 256, 0, stream>>>(featC, 32, 32, aggB, 16, 64,
                                             ws12[10].dst, 16, (const float*)d_in[25],
                                             ws12[11].dst, 32, 3, 16, (const float*)d_in[27],
                                             0, featC /*unused*/, popA);

  scene_sum_kernel<<<NSC, 256, 0, stream>>>(popA, bfull, sums);
  finalize_kernel<<<64, 256, 0, stream>>>(popA, bfull, sums, scnt, (float*)d_out);
}
